// FlaxGPT2Attention_3642132267671
// MI455X (gfx1250) — matrix operand order repaired and checked by reading, not compiled, not run on hardware
//
#include <hip/hip_runtime.h>

// ---------------------------------------------------------------------------
// Types for CDNA5 WMMA (wave32): D(16x16 f32) = A(16x32 bf16) x B(32x16 bf16) + C
// ---------------------------------------------------------------------------
typedef __attribute__((ext_vector_type(4)))  __bf16 v4bf;
typedef __attribute__((ext_vector_type(8)))  __bf16 v8bf;
typedef __attribute__((ext_vector_type(16))) __bf16 v16bf;
typedef __attribute__((ext_vector_type(8)))  float  v8f;

#define GK_B   2
#define GK_S   2048
#define GK_D   1024
#define GK_H   16
#define GK_HD  64
#define GK_M   (GK_B * GK_S)   // 4096 rows (b*s flattened)
#define GK_3D  (3 * GK_D)      // 3072

static __device__ __forceinline__ v16bf ld16(const __bf16* p) {
  v8bf lo = *(const v8bf*)p;
  v8bf hi = *(const v8bf*)(p + 8);
  return __builtin_shufflevector(lo, hi, 0,1,2,3,4,5,6,7,8,9,10,11,12,13,14,15);
}

static __device__ __forceinline__ v16bf ld8x2(const __bf16* p0, const __bf16* p1) {
  v8bf lo = *(const v8bf*)p0;
  v8bf hi = *(const v8bf*)p1;
  return __builtin_shufflevector(lo, hi, 0,1,2,3,4,5,6,7,8,9,10,11,12,13,14,15);
}

static __device__ __forceinline__ v8f wmma_bf16(v16bf a, v16bf b, v8f c) {
  return __builtin_amdgcn_wmma_f32_16x16x32_bf16(
      /*neg_a=*/false, a, /*neg_b=*/false, b,
      /*c_mod=*/(short)0, c, /*reuse_a=*/false, /*reuse_b=*/false);
}

static __device__ __forceinline__ v8f vzero8() {
  v8f r = {0.f, 0.f, 0.f, 0.f, 0.f, 0.f, 0.f, 0.f};
  return r;
}

// DPP16 row-rotate (16-lane row on wave32): pure-VALU cross-lane reductions.
template <int CTRL>
static __device__ __forceinline__ float dpp_rot(float x) {
  return __builtin_bit_cast(float,
      __builtin_amdgcn_update_dpp(0, __builtin_bit_cast(int, x),
                                  CTRL, 0xF, 0xF, true));
}
static __device__ __forceinline__ float rowmax16(float v) {
  v = fmaxf(v, dpp_rot<0x128>(v));  // row_ror:8
  v = fmaxf(v, dpp_rot<0x124>(v));  // row_ror:4
  v = fmaxf(v, dpp_rot<0x122>(v));  // row_ror:2
  v = fmaxf(v, dpp_rot<0x121>(v));  // row_ror:1
  return v;
}
static __device__ __forceinline__ float rowsum16(float v) {
  v += dpp_rot<0x128>(v);
  v += dpp_rot<0x124>(v);
  v += dpp_rot<0x122>(v);
  v += dpp_rot<0x121>(v);
  return v;
}

// Async 16B copy global -> LDS (GLOBAL_LOAD_ASYNC_TO_LDS_B128, ASYNCcnt).
static __device__ __forceinline__ void async_ld_b128(unsigned ldsoff, unsigned goff,
                                                     const void* base) {
  asm volatile("global_load_async_to_lds_b128 %0, %1, %2"
               :: "v"(ldsoff), "v"(goff), "s"(base) : "memory");
}
static __device__ __forceinline__ void wait_async0() {
  asm volatile("s_wait_asynccnt 0" ::: "memory");
}

// ---------------------------------------------------------------------------
// fp32 -> bf16 conversion (vectorized, grid-stride)
// ---------------------------------------------------------------------------
__global__ __launch_bounds__(256) void cvt_f32_to_bf16(
    const float* __restrict__ s, __bf16* __restrict__ d, int n) {
  int stride = (int)(gridDim.x * blockDim.x);
  for (int i = blockIdx.x * blockDim.x + threadIdx.x; i * 4 < n; i += stride) {
    float4 f = ((const float4*)s)[i];
    v4bf r = {(__bf16)f.x, (__bf16)f.y, (__bf16)f.z, (__bf16)f.w};
    ((v4bf*)d)[i] = r;
  }
}

// ---------------------------------------------------------------------------
// GEMM: C[M,N] = A[M,K] (bf16) @ W[N,K]^T (bf16) + bias[N], optional column
// scaling (used to fold the attention 1/sqrt(hd) into the Q columns).
//   Block tile 128x256, 8 waves (2Mx4N), each wave 64x64 = 16 WMMA / K=32.
//   Double-buffered LDS tiles filled with global_load_async_to_lds_b128:
//   issue async fill for tile i+1 after the barrier, compute tile i.
// ---------------------------------------------------------------------------
template <bool OUT_BF16>
__global__ __launch_bounds__(256) void gemm_wmma(
    const __bf16* __restrict__ A, const __bf16* __restrict__ W,
    const float* __restrict__ bias, void* __restrict__ Cout,
    int M, int N, int K, int qscale_cols) {
  const int bm   = blockIdx.y * 128;
  const int bn   = blockIdx.x * 256;
  const int tid  = threadIdx.x;
  const int lane = tid & 31;
  const int wid  = tid >> 5;
  const int half = lane >> 4;        // wave32 lane halves
  const int l16  = lane & 15;
  const int wm   = (wid >> 2) * 64;  // wave M offset: 0 or 64
  const int wn   = (wid & 3) * 64;   // wave N offset: 0..192

  // Rows padded to 40 elems (80B): bank spread, 16B alignment kept.
  __shared__ __align__(16) __bf16 As[2][128 * 40];
  __shared__ __align__(16) __bf16 Ws[2][256 * 40];

  auto stage = [&](int buf, int kt) {
#pragma unroll
    for (int cc = 0; cc < 2; ++cc) {
      int chunk = tid + cc * 256;
      int r = chunk >> 2;            // 0..127
      int c = (chunk & 3) * 8;       // 0,8,16,24
      async_ld_b128((unsigned)(size_t)&As[buf][r * 40 + c],
                    (unsigned)(((bm + r) * K + kt + c) * 2), A);
    }
#pragma unroll
    for (int cc = 0; cc < 4; ++cc) {
      int chunk = tid + cc * 256;
      int r = chunk >> 2;            // 0..255
      int c = (chunk & 3) * 8;
      async_ld_b128((unsigned)(size_t)&Ws[buf][r * 40 + c],
                    (unsigned)(((bn + r) * K + kt + c) * 2), W);
    }
  };

  v8f acc[4][4];
#pragma unroll
  for (int mi = 0; mi < 4; ++mi)
#pragma unroll
    for (int ni = 0; ni < 4; ++ni) acc[mi][ni] = vzero8();

  stage(0, 0);
  int buf = 0;
  for (int kt = 0; kt < K; kt += 32) {
    wait_async0();    // our copies for `buf` done (per-wave counter)...
    __syncthreads();  // ...barrier makes the whole buffer globally ready
    if (kt + 32 < K) stage(buf ^ 1, kt + 32);  // overlap fill with compute

    v16bf bfrag[4];
#pragma unroll
    for (int ni = 0; ni < 4; ++ni)
      bfrag[ni] = ld16(&Ws[buf][(wn + ni * 16 + l16) * 40 + half * 16]);
#pragma unroll
    for (int mi = 0; mi < 4; ++mi) {
      const __bf16* ap = &As[buf][(wm + mi * 16 + l16) * 40];
      v16bf afrag = ld16(ap + half * 16);
#pragma unroll
      for (int ni = 0; ni < 4; ++ni)
        acc[mi][ni] = wmma_bf16(afrag, bfrag[ni], acc[mi][ni]);
    }
    buf ^= 1;
  }

  // Epilogue: C layout — VGPR j, lanes 0-15: row j; lanes 16-31: row j+8.
#pragma unroll
  for (int mi = 0; mi < 4; ++mi)
#pragma unroll
    for (int ni = 0; ni < 4; ++ni) {
      int col = bn + wn + ni * 16 + l16;
      float bv = bias[col];
      float sc = (col < qscale_cols) ? 0.125f : 1.0f;  // fold 1/sqrt(64) into Q
#pragma unroll
      for (int j = 0; j < 8; ++j) {
        int row = bm + wm + mi * 16 + j + half * 8;
        float v = (acc[mi][ni][j] + bv) * sc;
        if (OUT_BF16)
          ((__bf16*)Cout)[(size_t)row * N + col] = (__bf16)v;
        else
          ((float*)Cout)[(size_t)row * N + col] = v;
      }
    }
}

// ---------------------------------------------------------------------------
// Flash attention: block = (b, h, 128-query tile); 8 waves x 16 query rows.
// Double-buffered KV staging: K via async copy, V prefetched into registers
// one tile ahead and transposed into the other LDS buffer during compute.
// Online softmax in fp32 with DPP row reductions; Q pre-scaled in QKV GEMM.
// ---------------------------------------------------------------------------
__global__ __launch_bounds__(256) void attn_flash(
    const __bf16* __restrict__ qkv, __bf16* __restrict__ out) {
  const int qb   = blockIdx.x * 128;
  const int h    = blockIdx.y;
  const int b    = blockIdx.z;
  const int tid  = threadIdx.x;
  const int lane = tid & 31;
  const int wid  = tid >> 5;
  const int half = lane >> 4;
  const int l16  = lane & 15;

  __shared__ __align__(16) __bf16 Kt[2][32 * 72];   // [key][hd], pad 72
  __shared__ __align__(16) __bf16 VtT[2][64 * 40];  // [hd][key], pad 40
  __shared__ __align__(16) __bf16 Pws[8][16 * 40];  // per-wave P tile, pad 40

  const int q0w = qb + wid * 16;
  const int rs  = GK_3D;
  const __bf16* base = qkv + (size_t)b * GK_S * rs;

  const int sr = tid >> 3;           // staging: key row 0..31
  const int sc = (tid & 7) * 8;      // staging: hd 0..56

  auto stageK = [&](int buf, int kt) {
    async_ld_b128((unsigned)(size_t)&Kt[buf][sr * 72 + sc],
                  (unsigned)(((kt + sr) * rs + GK_D + h * GK_HD + sc) * 2), base);
  };
  auto loadV = [&](int kt) -> v8bf {
    return *(const v8bf*)(base + (size_t)(kt + sr) * rs + 2 * GK_D + h * GK_HD + sc);
  };
  auto storeV = [&](int buf, v8bf vv) {
#pragma unroll
    for (int i = 0; i < 8; ++i) VtT[buf][(sc + i) * 40 + sr] = vv[i];
  };

  // Q fragments (A operand), loaded once (already scaled by 1/sqrt(64)).
  v16bf qf[2];
  {
    const __bf16* qp = base + (size_t)(q0w + l16) * rs + h * GK_HD;
#pragma unroll
    for (int kb = 0; kb < 2; ++kb)
      qf[kb] = ld16(qp + kb * 32 + half * 16);
  }

  v8f oacc[4];
#pragma unroll
  for (int t = 0; t < 4; ++t) oacc[t] = vzero8();
  float mrow[8], lrow[8];
#pragma unroll
  for (int j = 0; j < 8; ++j) { mrow[j] = -3.0e38f; lrow[j] = 0.f; }

  const int kend = qb + 128;
  // Prologue: tile 0 staged, tile-1 V prefetched into registers.
  stageK(0, 0);
  storeV(0, loadV(0));
  v8bf vv = loadV(32 < kend - 32 ? 32 : kend - 32);

  int buf = 0;
  for (int kt = 0; kt < kend; kt += 32) {
    wait_async0();    // our async K copies for `buf` done
    __syncthreads();  // buffer globally ready; prev readers of buf^1 done
    if (kt + 32 < kend) {
      stageK(buf ^ 1, kt + 32);   // async fill next K tile
      storeV(buf ^ 1, vv);        // transpose prefetched V into next buffer
      int ktn = kt + 64; if (ktn > kend - 32) ktn = kend - 32;
      vv = loadV(ktn);            // prefetch V two tiles ahead
    }

    if (kt <= q0w + 15) {  // tile not fully masked for this wave
      // --- scores: 16 queries x 32 keys, two 16x16 C tiles ---
      v8f s0 = vzero8(), s1 = vzero8();
      s0 = wmma_bf16(qf[0], ld16(&Kt[buf][l16 * 72 + half * 16]), s0);
      s0 = wmma_bf16(qf[1], ld16(&Kt[buf][l16 * 72 + 32 + half * 16]), s0);
      s1 = wmma_bf16(qf[0], ld16(&Kt[buf][(16 + l16) * 72 + half * 16]), s1);
      s1 = wmma_bf16(qf[1], ld16(&Kt[buf][(16 + l16) * 72 + 32 + half * 16]), s1);

      const bool needMask = (kt + 31 > q0w);  // only diagonal tiles
#pragma unroll
      for (int j = 0; j < 8; ++j) {
        float a  = s0[j];
        float c2 = s1[j];
        if (needMask) {
          int qr = q0w + j + half * 8;
          if (kt + l16 > qr)      a  = -3.0e38f;
          if (kt + 16 + l16 > qr) c2 = -3.0e38f;
        }
        float mx    = rowmax16(fmaxf(a, c2));
        float mnew  = fmaxf(mrow[j], mx);
        float alpha = __expf(mrow[j] - mnew);
        float p0 = __expf(a - mnew);
        float p1 = __expf(c2 - mnew);
        s0[j] = p0; s1[j] = p1;
        float rsum = rowsum16(p0 + p1);
        lrow[j] = lrow[j] * alpha + rsum;
        mrow[j] = mnew;
        oacc[0][j] *= alpha; oacc[1][j] *= alpha;
        oacc[2][j] *= alpha; oacc[3][j] *= alpha;
      }

      // --- P: C layout -> LDS (row-major 16x32 bf16) -> A layout ---
#pragma unroll
      for (int j = 0; j < 8; ++j) {
        int row = j + half * 8;
        Pws[wid][row * 40 + l16]      = (__bf16)s0[j];
        Pws[wid][row * 40 + 16 + l16] = (__bf16)s1[j];
      }
      asm volatile("s_wait_dscnt 0" ::: "memory");  // intra-wave LDS RAW
      const __bf16* pp = &Pws[wid][l16 * 40];
      v16bf pf = ld16(pp + half * 16);

      // --- O += P(16x32) @ V(32x16 per hd tile); V^T: contiguous LDS reads ---
#pragma unroll
      for (int t = 0; t < 4; ++t) {
        v16bf vf = ld16(&VtT[buf][(t * 16 + l16) * 40 + half * 16]);
        oacc[t] = wmma_bf16(pf, vf, oacc[t]);
      }
    }
    buf ^= 1;
  }

  // Normalize (one reciprocal per row) and store bf16 output.
#pragma unroll
  for (int j = 0; j < 8; ++j) {
    int qr = q0w + j + half * 8;
    float inv = 1.0f / lrow[j];
#pragma unroll
    for (int t = 0; t < 4; ++t) {
      int hd = t * 16 + l16;
      out[(size_t)(b * GK_S + qr) * GK_D + h * GK_HD + hd] =
          (__bf16)(oacc[t][j] * inv);
    }
  }
}

// ---------------------------------------------------------------------------
// Launcher
// ---------------------------------------------------------------------------
extern "C" void kernel_launch(void* const* d_in, const int* in_sizes, int n_in,
                              void* d_out, int out_size, void* d_ws, size_t ws_size,
                              hipStream_t stream) {
  (void)in_sizes; (void)n_in; (void)out_size; (void)ws_size;
  const float* x      = (const float*)d_in[0];  // [2,2048,1024]
  const float* w_attn = (const float*)d_in[1];  // [3072,1024]
  const float* b_attn = (const float*)d_in[2];  // [3072]
  const float* w_proj = (const float*)d_in[3];  // [1024,1024]
  const float* b_proj = (const float*)d_in[4];  // [1024]
  float* out = (float*)d_out;                   // [2,2048,1024] fp32

  char* ws = (char*)d_ws;
  size_t o = 0;
  __bf16* xb    = (__bf16*)(ws + o); o += (size_t)GK_M  * GK_D  * 2;  //  8 MiB
  __bf16* wqb   = (__bf16*)(ws + o); o += (size_t)GK_3D * GK_D  * 2;  //  6 MiB
  __bf16* wpb   = (__bf16*)(ws + o); o += (size_t)GK_D  * GK_D  * 2;  //  2 MiB
  __bf16* qkvb  = (__bf16*)(ws + o); o += (size_t)GK_M  * GK_3D * 2;  // 24 MiB
  __bf16* attnb = (__bf16*)(ws + o); o += (size_t)GK_M  * GK_D  * 2;  //  8 MiB

  // 1) fp32 -> bf16 conversions
  cvt_f32_to_bf16<<<dim3(1024), dim3(256), 0, stream>>>(x,      xb,  GK_M  * GK_D);
  cvt_f32_to_bf16<<<dim3(768),  dim3(256), 0, stream>>>(w_attn, wqb, GK_3D * GK_D);
  cvt_f32_to_bf16<<<dim3(256),  dim3(256), 0, stream>>>(w_proj, wpb, GK_D  * GK_D);

  // 2) fused QKV projection (Q columns pre-scaled by 1/sqrt(64)) -> bf16
  gemm_wmma<true><<<dim3(GK_3D / 256, GK_M / 128), dim3(256), 0, stream>>>(
      xb, wqb, b_attn, (void*)qkvb, GK_M, GK_3D, GK_D, GK_D);

  // 3) causal flash attention -> bf16 [4096,1024]
  attn_flash<<<dim3(GK_S / 128, GK_H, GK_B), dim3(256), 0, stream>>>(qkvb, attnb);

  // 4) output projection: [4096,1024] x [1024,1024]^T -> fp32 d_out
  gemm_wmma<false><<<dim3(GK_D / 256, GK_M / 128), dim3(256), 0, stream>>>(
      attnb, wpb, b_proj, (void*)out, GK_M, GK_D, GK_D, 0);
}
